// Convolution_66726611910944
// MI455X (gfx1250) — matrix operand order repaired and checked by reading, not compiled, not run on hardware
//
#include <hip/hip_runtime.h>
#include <cstdint>

typedef __attribute__((ext_vector_type(16))) _Float16 v16h;
typedef __attribute__((ext_vector_type(8)))  float    v8f;

#define NUM_NODES 10000
#define NUM_EDGES 100000
#define EDGES_PER_BLOCK 16
#define NBLOCKS (NUM_EDGES / EDGES_PER_BLOCK)   // 6250, exact

// constants
#define INV_SQRT3   0.57735026918962576f
#define PATH_NORM   0.17677669529663689f   // 1/sqrt(32)
#define INV_SQRT10  0.31622776601683794f
#define GEMM_SCALE  0.0625f                // 1/sqrt(256)

union FragU {
    uint4 q[2];
    v16h  v;
};

// ---------------------------------------------------------------------------
// Pre-swizzle W2 (fp32 [256,2048]) into f16 B-fragment order in workspace.
// Layout: [ntile=128][kstep=8][lane=32][8 dwords], so each lane's fragment
// is 32 contiguous bytes (two b128 loads), and a wave's access is a fully
// coalesced 1KB stream.
// B-matrix (32x16) per-lane layout: dword j holds K = ks*32 + 2j + (lane>=16?16:0)
// (low half) and K+1 (high half), N = ntile*16 + (lane&15).
// ---------------------------------------------------------------------------
__global__ void prep_w2_kernel(const float* __restrict__ W2,
                               uint32_t* __restrict__ w2p) {
    int idx = blockIdx.x * 256 + threadIdx.x;   // 0 .. 262143
    if (idx >= 128 * 8 * 32 * 8) return;
    int j     = idx & 7;
    int lane  = (idx >> 3) & 31;
    int ks    = (idx >> 8) & 7;
    int ntile = idx >> 11;
    int n  = ntile * 16 + (lane & 15);
    int k0 = ks * 32 + 2 * j + ((lane & 16) ? 16 : 0);
    union { _Float16 h[2]; uint32_t u; } p;
    p.h[0] = (_Float16)W2[(size_t)k0 * 2048 + n];
    p.h[1] = (_Float16)W2[(size_t)(k0 + 1) * 2048 + n];
    w2p[idx] = p.u;
}

// ---------------------------------------------------------------------------
// Fused: h = relu(scal@W1/sqrt3) -> weight = h@W2/16 (f16 WMMA, fp32 acc)
//        -> e3nn tensor product -> atomic scatter to nodes.
// 16 edges/block, 512 threads = 16 waves. GEMM done in two N=1024 halves so
// the f16 weight tile fits in LDS next to the h tile.
// ---------------------------------------------------------------------------
__global__ __launch_bounds__(512)
void conv_fused_kernel(const float* __restrict__ nodef,
                       const int*   __restrict__ esrc,
                       const int*   __restrict__ edst,
                       const float* __restrict__ eattr,
                       const float* __restrict__ escal,
                       const float* __restrict__ W1,
                       const uint4* __restrict__ w2p,
                       float*       __restrict__ out) {
    __shared__ __align__(16) _Float16 h_lds[16 * 256];    // A operand, f16
    __shared__ __align__(16) _Float16 w_lds[16 * 1024];   // weight half-tile, f16
    __shared__ float x_lds[16 * 64];                      // gathered node feats
    __shared__ float sh_lds[16 * 4];                      // edge_attr
    __shared__ int   dst_lds[16];

    const int tid  = threadIdx.x;
    const int lane = tid & 31;
    const int wave = tid >> 5;
    const int e0   = blockIdx.x * EDGES_PER_BLOCK;

    // ---- stage edge data ----------------------------------------------------
    for (int i = tid; i < 16 * 64; i += 512) {
        int e = i >> 6, c = i & 63;
        x_lds[i] = nodef[(size_t)esrc[e0 + e] * 64 + c];
    }
    if (tid < 64) sh_lds[tid] = eattr[(size_t)e0 * 4 + tid];
    if (tid < 16) dst_lds[tid] = edst[e0 + tid];

    // ---- h = relu(edge_scalars @ W1 / sqrt3), stored f16 row-major ---------
    for (int i = tid; i < 16 * 256; i += 512) {
        int e = i >> 8, c = i & 255;
        const float* es = &escal[(size_t)(e0 + e) * 3];
        float v = (es[0] * W1[c] + es[1] * W1[256 + c] + es[2] * W1[512 + c]) * INV_SQRT3;
        h_lds[i] = (_Float16)(v > 0.0f ? v : 0.0f);
    }
    __syncthreads();

    const int m      = lane & 15;           // A-row / C-col index
    const int hiHalf = (lane >= 16) ? 1 : 0;

    for (int half = 0; half < 2; ++half) {
        // ---- GEMM: 16 x 1024 slab of weight = h @ W2 ------------------------
        v8f acc[4];
#pragma unroll
        for (int t = 0; t < 4; ++t)
#pragma unroll
            for (int v = 0; v < 8; ++v) acc[t][v] = 0.0f;

#pragma unroll
        for (int ks = 0; ks < 8; ++ks) {
            // A fragment (16x32 f16): contiguous K = ks*32 + 16*hiHalf + s, matching B
            int kbase = ks * 32 + (hiHalf ? 16 : 0);
            FragU A;
            A.q[0] = *(const uint4*)&h_lds[m * 256 + kbase];
            A.q[1] = *(const uint4*)&h_lds[m * 256 + kbase + 8];
#pragma unroll
            for (int t = 0; t < 4; ++t) {
                int ntile = half * 64 + wave * 4 + t;
                const uint4* bptr = &w2p[(((size_t)ntile * 8 + ks) * 32 + lane) * 2];
                FragU B;
                B.q[0] = bptr[0];
                B.q[1] = bptr[1];
                acc[t] = __builtin_amdgcn_wmma_f32_16x16x32_f16(
                    false, A.v, false, B.v, (short)0, acc[t], false, false);
            }
        }

        // ---- dump slab to LDS as f16 (C/D layout: M = v + 8*hiHalf) --------
#pragma unroll
        for (int t = 0; t < 4; ++t) {
            int col = wave * 64 + t * 16 + m;
#pragma unroll
            for (int v = 0; v < 8; ++v)
                w_lds[(hiHalf * 8 + v) * 1024 + col] =
                    (_Float16)(acc[t][v] * GEMM_SCALE);
        }
        __syncthreads();

        // ---- tensor product on this half -----------------------------------
        const int e = tid >> 5;        // edge 0..15
        const int w = tid & 31;        // output mul 0..31
        const float* xe  = &x_lds[e * 64];
        const float* she = &sh_lds[e * 4];
        const float  s0  = she[0];
        const int    dO  = dst_lds[e] * 128;

        if (half == 0) {
            // o0[w] = pn * s0 * sum_u x0[u]*w1[u,w]
            //       + pn/sqrt3 * sum_u (x1[u]·s1)*w2[u,w]
            float a0 = 0.0f, a0b = 0.0f;
#pragma unroll
            for (int u = 0; u < 16; ++u) {
                a0 += xe[u] * (float)w_lds[e * 1024 + u * 32 + w];
                float c1 = xe[16 + u * 3] * she[1] + xe[16 + u * 3 + 1] * she[2] +
                           xe[16 + u * 3 + 2] * she[3];
                a0b += c1 * (float)w_lds[e * 1024 + 512 + u * 32 + w];
            }
            float o = (PATH_NORM * s0) * a0 + (PATH_NORM * INV_SQRT3) * a0b;
            atomicAdd(&out[dO + w], o * INV_SQRT10);
        } else {
            // o1[w,j] = pn * ( s1[j] * sum_u x0[u]*w3[u,w]
            //                + s0    * sum_u x1[u,j]*w4[u,w] )
            float y3 = 0.0f, z0 = 0.0f, z1 = 0.0f, z2 = 0.0f;
#pragma unroll
            for (int u = 0; u < 16; ++u) {
                float w3v = (float)w_lds[e * 1024 + u * 32 + w];
                float w4v = (float)w_lds[e * 1024 + 512 + u * 32 + w];
                y3 += xe[u] * w3v;
                z0 += xe[16 + u * 3 + 0] * w4v;
                z1 += xe[16 + u * 3 + 1] * w4v;
                z2 += xe[16 + u * 3 + 2] * w4v;
            }
            float o0j = PATH_NORM * (she[1] * y3 + s0 * z0);
            float o1j = PATH_NORM * (she[2] * y3 + s0 * z1);
            float o2j = PATH_NORM * (she[3] * y3 + s0 * z2);
            atomicAdd(&out[dO + 32 + w * 3 + 0], o0j * INV_SQRT10);
            atomicAdd(&out[dO + 32 + w * 3 + 1], o1j * INV_SQRT10);
            atomicAdd(&out[dO + 32 + w * 3 + 2], o2j * INV_SQRT10);
        }
        __syncthreads();   // protect w_lds before next half overwrites it
    }
}

extern "C" void kernel_launch(void* const* d_in, const int* in_sizes, int n_in,
                              void* d_out, int out_size, void* d_ws, size_t ws_size,
                              hipStream_t stream) {
    (void)in_sizes; (void)n_in; (void)ws_size;
    const float* nodef = (const float*)d_in[0];
    const int*   esrc  = (const int*)d_in[1];
    const int*   edst  = (const int*)d_in[2];
    const float* eattr = (const float*)d_in[3];
    const float* escal = (const float*)d_in[4];
    const float* W1    = (const float*)d_in[5];
    const float* W2    = (const float*)d_in[6];
    float* out = (float*)d_out;

    // output accumulated via atomics -> zero it first
    hipMemsetAsync(d_out, 0, (size_t)out_size * sizeof(float), stream);

    // pre-swizzle W2 into f16 B-fragment order (1 MB in d_ws)
    prep_w2_kernel<<<1024, 256, 0, stream>>>(W2, (uint32_t*)d_ws);

    conv_fused_kernel<<<NBLOCKS, 512, 0, stream>>>(
        nodef, esrc, edst, eattr, escal, W1, (const uint4*)d_ws, out);
}
